// VectorQuantizer_69715909149317
// MI455X (gfx1250) — compile-verified, matrix operand order checked by reading, not run on hardware
//
#include <hip/hip_runtime.h>

typedef __attribute__((ext_vector_type(16))) __bf16 v16bf;
typedef __attribute__((ext_vector_type(8)))  __bf16 v8bf;
typedef __attribute__((ext_vector_type(8)))  float  v8f;

#define M_ROWS   32768      // 16*2048
#define D_DIM    512
#define N_CODES  8192
#define ZQ_ELEMS 16777216   // 32768*512

// ---- workspace layout (bytes) ----
#define WS_EBF      0            // bf16 embedding: 8192*512*2 = 8388608
#define WS_ENORM    8388608      // f32 ||e||^2:     8192*4    = 32768
#define WS_CODES    8421376      // i32 codes:       32768*4   = 131072
#define WS_COUNTS   8552448      // u32 counts:      8192*4    = 32768
#define WS_PARTIAL  8585216      // f32 partials:    32768*4   = 131072

__device__ __forceinline__ unsigned short f32_to_bf16_bits(float f) {
    unsigned u = __builtin_bit_cast(unsigned, f);
    u = u + 0x7FFFu + ((u >> 16) & 1u);   // round-to-nearest-even
    return (unsigned short)(u >> 16);
}
__device__ __forceinline__ __bf16 f32_to_bf16(float f) {
    unsigned short s = f32_to_bf16_bits(f);
    return __builtin_bit_cast(__bf16, s);
}

// ---------------------------------------------------------------------------
// Kernel 1: embedding f32 -> bf16, ||e||^2, zero counts.  8192 blocks x 128.
// ---------------------------------------------------------------------------
__global__ void __launch_bounds__(128)
vq_prep_kernel(const float* __restrict__ emb, unsigned short* __restrict__ ebf,
               float* __restrict__ enorm, unsigned* __restrict__ counts) {
    const int row = blockIdx.x;
    const int t   = threadIdx.x;
    const float4 v = *(const float4*)(emb + (size_t)row * D_DIM + t * 4);
    ushort4 b;
    b.x = f32_to_bf16_bits(v.x); b.y = f32_to_bf16_bits(v.y);
    b.z = f32_to_bf16_bits(v.z); b.w = f32_to_bf16_bits(v.w);
    *(ushort4*)(ebf + (size_t)row * D_DIM + t * 4) = b;

    float ss = v.x * v.x + v.y * v.y + v.z * v.z + v.w * v.w;
    __shared__ float red[128];
    red[t] = ss; __syncthreads();
    for (int k = 64; k > 0; k >>= 1) { if (t < k) red[t] += red[t + k]; __syncthreads(); }
    if (t == 0) { enorm[row] = red[0]; counts[row] = 0u; }
}

// ---------------------------------------------------------------------------
// Kernel 2: WMMA argmin.  256 blocks x 256 threads (8 waves).
// Each wave: 16 z-rows; loops over 256 tiles of 32 codes; B panel in LDS.
// Two accumulator chains + 1-deep B-fragment software pipeline so ds_load
// latency overlaps the matrix pipe.
// ---------------------------------------------------------------------------
__global__ void __launch_bounds__(256)
vq_argmin_kernel(const float* __restrict__ z, const unsigned short* __restrict__ ebf_u,
                 const float* __restrict__ enorm, int* __restrict__ codes) {
    __shared__ __align__(16) unsigned short ldsB[32 * D_DIM];   // 32 KB

    const int t    = threadIdx.x;
    const int wave = t >> 5;
    const int lane = t & 31;
    const int hi   = (lane >> 4) & 1;     // K-half select per ISA fragment layout
    const int col  = lane & 15;           // A: M row within tile / B: code column
    const int rowBase = (blockIdx.x * 8 + wave) * 16;

    // --- load & convert A fragments once: 16 rows x 512 K, bf16, ISA layout ---
    v16bf afrag[16];
    const float* zr = z + (size_t)(rowBase + col) * D_DIM;
    #pragma unroll
    for (int kk = 0; kk < 16; ++kk) {
        const int k0 = kk * 32;
        const float4 a0 = *(const float4*)(zr + k0 + hi * 8);
        const float4 a1 = *(const float4*)(zr + k0 + hi * 8 + 4);
        const float4 a2 = *(const float4*)(zr + k0 + 16 + hi * 8);
        const float4 a3 = *(const float4*)(zr + k0 + 16 + hi * 8 + 4);
        afrag[kk][0]  = f32_to_bf16(a0.x); afrag[kk][1]  = f32_to_bf16(a0.y);
        afrag[kk][2]  = f32_to_bf16(a0.z); afrag[kk][3]  = f32_to_bf16(a0.w);
        afrag[kk][4]  = f32_to_bf16(a1.x); afrag[kk][5]  = f32_to_bf16(a1.y);
        afrag[kk][6]  = f32_to_bf16(a1.z); afrag[kk][7]  = f32_to_bf16(a1.w);
        afrag[kk][8]  = f32_to_bf16(a2.x); afrag[kk][9]  = f32_to_bf16(a2.y);
        afrag[kk][10] = f32_to_bf16(a2.z); afrag[kk][11] = f32_to_bf16(a2.w);
        afrag[kk][12] = f32_to_bf16(a3.x); afrag[kk][13] = f32_to_bf16(a3.y);
        afrag[kk][14] = f32_to_bf16(a3.z); afrag[kk][15] = f32_to_bf16(a3.w);
    }

    float bestS[8]; int bestI[8];
    #pragma unroll
    for (int r = 0; r < 8; ++r) { bestS[r] = 3.4e38f; bestI[r] = 0; }

    const int cr0 = col;        // LDS row for chain 0 (codes n0+0..15)
    const int cr1 = col + 16;   // LDS row for chain 1 (codes n0+16..31)

    for (int n0 = 0; n0 < N_CODES; n0 += 32) {
        __syncthreads();
        // stage 32 codes x 512 K bf16 (contiguous 32KB) into LDS, 256 threads
        const uint4* src = (const uint4*)(ebf_u + (size_t)n0 * D_DIM);
        uint4* dst = (uint4*)ldsB;
        #pragma unroll
        for (int i = 0; i < 8; ++i) dst[t + i * 256] = src[t + i * 256];
        __syncthreads();

        v8f acc0 = {};
        v8f acc1 = {};

        // prologue: fragment kk=0 for both chains
        v8bf l0, h0, l1, h1;
        {
            const __bf16* b0 = (const __bf16*)(ldsB + cr0 * D_DIM + hi * 8);
            const __bf16* b1 = (const __bf16*)(ldsB + cr1 * D_DIM + hi * 8);
            l0 = *(const v8bf*)b0; h0 = *(const v8bf*)(b0 + 16);
            l1 = *(const v8bf*)b1; h1 = *(const v8bf*)(b1 + 16);
        }
        #pragma unroll
        for (int kk = 0; kk < 16; ++kk) {
            v8bf nl0, nh0, nl1, nh1;
            if (kk < 15) {   // prefetch next K-step while WMMAs execute
                const int k0 = (kk + 1) * 32;
                const __bf16* b0 = (const __bf16*)(ldsB + cr0 * D_DIM + k0 + hi * 8);
                const __bf16* b1 = (const __bf16*)(ldsB + cr1 * D_DIM + k0 + hi * 8);
                nl0 = *(const v8bf*)b0; nh0 = *(const v8bf*)(b0 + 16);
                nl1 = *(const v8bf*)b1; nh1 = *(const v8bf*)(b1 + 16);
            }
            const v16bf bf0 = __builtin_shufflevector(l0, h0,
                0,1,2,3,4,5,6,7,8,9,10,11,12,13,14,15);
            const v16bf bf1 = __builtin_shufflevector(l1, h1,
                0,1,2,3,4,5,6,7,8,9,10,11,12,13,14,15);
            acc0 = __builtin_amdgcn_wmma_f32_16x16x32_bf16(
                false, afrag[kk], false, bf0, (short)0, acc0, false, false);
            acc1 = __builtin_amdgcn_wmma_f32_16x16x32_bf16(
                false, afrag[kk], false, bf1, (short)0, acc1, false, false);
            l0 = nl0; h0 = nh0; l1 = nl1; h1 = nh1;
        }

        const float en0 = enorm[n0 + col];
        const float en1 = enorm[n0 + 16 + col];
        const int   c0  = n0 + col;
        const int   c1  = n0 + 16 + col;
        #pragma unroll
        for (int r = 0; r < 8; ++r) {      // lower codes first: keeps first-index tie-break
            const float s0 = en0 - 2.0f * acc0[r];
            if (s0 < bestS[r]) { bestS[r] = s0; bestI[r] = c0; }
            const float s1 = en1 - 2.0f * acc1[r];
            if (s1 < bestS[r]) { bestS[r] = s1; bestI[r] = c1; }
        }
    }

    // cross-lane argmin within each 16-lane group (C layout: lanes = N, VGPRs = M)
    #pragma unroll
    for (int m = 1; m < 16; m <<= 1) {
        #pragma unroll
        for (int r = 0; r < 8; ++r) {
            const float os = __shfl_xor(bestS[r], m, 32);
            const int   oi = __shfl_xor(bestI[r], m, 32);
            if (os < bestS[r] || (os == bestS[r] && oi < bestI[r])) {
                bestS[r] = os; bestI[r] = oi;
            }
        }
    }
    if (col == 0) {
        const int mb = rowBase + hi * 8;
        #pragma unroll
        for (int r = 0; r < 8; ++r) codes[mb + r] = bestI[r];
    }
}

// ---------------------------------------------------------------------------
// Kernel 3: gather z_q (== z_q_st), codes-as-float, per-row diff^2 partials,
// exact integer counts.  32768 blocks x 256 threads.
// ---------------------------------------------------------------------------
__global__ void __launch_bounds__(256)
vq_gather_kernel(const float* __restrict__ z, const float* __restrict__ emb,
                 const int* __restrict__ codes, unsigned* __restrict__ counts,
                 float* __restrict__ partials, float* __restrict__ out) {
    const int m = blockIdx.x;
    const int t = threadIdx.x;
    const int code = codes[m];
    float s = 0.0f;
    for (int d = t; d < D_DIM; d += 256) {
        const float e  = emb[(size_t)code * D_DIM + d];
        const float zv = z[(size_t)m * D_DIM + d];
        out[(size_t)m * D_DIM + d] = e;
        const float df = e - zv;
        s += df * df;
    }
    __shared__ float red[256];
    red[t] = s; __syncthreads();
    for (int k = 128; k > 0; k >>= 1) { if (t < k) red[t] += red[t + k]; __syncthreads(); }
    if (t == 0) {
        partials[m] = red[0];
        atomicAdd(&counts[code], 1u);
        out[(size_t)ZQ_ELEMS + m] = (float)code;
    }
}

// ---------------------------------------------------------------------------
// Kernel 4: deterministic final reduction -> loss & perplexity. 1 block x 256.
// ---------------------------------------------------------------------------
__global__ void __launch_bounds__(256)
vq_final_kernel(const unsigned* __restrict__ counts, const float* __restrict__ partials,
                float* __restrict__ out) {
    const int t = threadIdx.x;
    __shared__ float red[256];

    float s = 0.0f;
    for (int i = t; i < M_ROWS; i += 256) s += partials[i];   // fixed order
    red[t] = s; __syncthreads();
    for (int k = 128; k > 0; k >>= 1) { if (t < k) red[t] += red[t + k]; __syncthreads(); }
    const float sumd2 = red[0];
    __syncthreads();

    float ent = 0.0f;
    for (int c = t; c < N_CODES; c += 256) {
        const float p = (float)counts[c] * (1.0f / (float)M_ROWS);
        ent += p * logf(p + 1e-10f);
    }
    red[t] = ent; __syncthreads();
    for (int k = 128; k > 0; k >>= 1) { if (t < k) red[t] += red[t + k]; __syncthreads(); }
    if (t == 0) {
        out[(size_t)ZQ_ELEMS + M_ROWS]     = 0.25f * sumd2 / (float)ZQ_ELEMS;  // loss
        out[(size_t)ZQ_ELEMS + M_ROWS + 1] = expf(-red[0]);                    // perplexity
    }
}

extern "C" void kernel_launch(void* const* d_in, const int* in_sizes, int n_in,
                              void* d_out, int out_size, void* d_ws, size_t ws_size,
                              hipStream_t stream) {
    const float* z   = (const float*)d_in[0];   // [16,2048,512] f32
    const float* emb = (const float*)d_in[1];   // [8192,512]    f32
    float* out = (float*)d_out;

    char* ws = (char*)d_ws;
    unsigned short* ebf   = (unsigned short*)(ws + WS_EBF);
    float*          enorm = (float*)(ws + WS_ENORM);
    int*            codes = (int*)(ws + WS_CODES);
    unsigned*       cnts  = (unsigned*)(ws + WS_COUNTS);
    float*          parts = (float*)(ws + WS_PARTIAL);

    vq_prep_kernel<<<N_CODES, 128, 0, stream>>>(emb, ebf, enorm, cnts);
    vq_argmin_kernel<<<M_ROWS / 128, 256, 0, stream>>>(z, ebf, enorm, codes);
    vq_gather_kernel<<<M_ROWS, 256, 0, stream>>>(z, emb, codes, cnts, parts, out);
    vq_final_kernel<<<1, 256, 0, stream>>>(cnts, parts, out);
}
